// LandslideEEGMoE_61615600828923
// MI455X (gfx1250) — compile-verified, matrix operand order checked by reading, not run on hardware
//
#include <hip/hip_runtime.h>

// ---------------- problem constants ----------------
#define L_     4
#define D_     128
#define HID_   512
#define NH_    4
#define ES_    6
#define ESH_   2
#define NC_    2
#define FLAT_  648
#define FLATP_ 672      // FLAT padded to multiple of 32 for WMMA K loop
#define ENCH_  64
#define B_     256
#define T_     100
#define NTOK_  (B_ * T_)   // 25600, multiple of 256
#define LDST_  (HID_ + 8)  // padded LDS row stride (bank-conflict avoidance)

// ---------------- vector types ----------------
typedef short  v8s  __attribute__((ext_vector_type(8)));
typedef __bf16 v16bf __attribute__((ext_vector_type(16)));
typedef float  v8f  __attribute__((ext_vector_type(8)));

__device__ __forceinline__ unsigned short f2bf(float f) {
  unsigned int u = __float_as_uint(f);
  u += 0x7FFFu + ((u >> 16) & 1u);     // round-to-nearest-even
  return (unsigned short)(u >> 16);
}

__device__ __forceinline__ v16bf ldfrag(const unsigned short* p) {
  v8s lo = *(const v8s*)p;
  v8s hi = *(const v8s*)(p + 16);
  return __builtin_bit_cast(
      v16bf, __builtin_shufflevector(lo, hi, 0, 1, 2, 3, 4, 5, 6, 7, 8, 9, 10,
                                     11, 12, 13, 14, 15));
}

#define WMMA_BF16(ACCV, AV, BV)                                             \
  ACCV = __builtin_amdgcn_wmma_f32_16x16x32_bf16(false, AV, false, BV,      \
                                                 (short)0, ACCV, false, false)

__device__ __forceinline__ float gelu_exact(float v) {
  return 0.5f * v * (1.f + erff(v * 0.70710678118654752f));
}

// ---------------- fp32 -> bf16 convert with optional K padding ----------------
__global__ void cvt_pad_bf16_kernel(const float* __restrict__ src,
                                    unsigned short* __restrict__ dst,
                                    int M, int K, int Kp) {
  long idx = (long)blockIdx.x * blockDim.x + threadIdx.x;
  long tot = (long)M * Kp;
  if (idx >= tot) return;
  int  k = (int)(idx % Kp);
  long m = idx / Kp;
  dst[idx] = (k < K) ? f2bf(src[m * (long)K + k]) : (unsigned short)0;
}

// ---------------- WMMA bf16 GEMM: C[M,N] = act(A[M,K] @ W[N,K]^T + bias) ----
// M % 256 == 0, N % 64 == 0, K % 32 == 0. 8 waves; block tile 256x64;
// wave tile 32x64 (2 A-frags x 4 B-frags -> 8 WMMA per 32-deep K step).
// ACT: 0 none, 1 relu, 2 gelu. OUTMODE: 0 f32 store, 1 bf16 store.
template <int ACT, int OUTMODE>
__global__ __launch_bounds__(256) void gemm_bf16_wmma(
    const unsigned short* __restrict__ A, const unsigned short* __restrict__ W,
    const float* __restrict__ bias, void* __restrict__ Cout,
    int M, int N, int K) {
  const int lane = threadIdx.x & 31;
  const int wave = threadIdx.x >> 5;
  const int l16  = lane & 15;
  const int hi   = lane >> 4;          // K sub-chunk select per ISA 7.12.2
  const int m0   = blockIdx.y * 256 + wave * 32;
  const int n0   = blockIdx.x * 64;

  v8f acc[2][4] = {};

  const unsigned short* ap0 = A + (size_t)(m0 + l16) * K + hi * 8;
  const unsigned short* ap1 = A + (size_t)(m0 + 16 + l16) * K + hi * 8;
  const unsigned short* wp0 = W + (size_t)(n0 +  0 + l16) * K + hi * 8;
  const unsigned short* wp1 = W + (size_t)(n0 + 16 + l16) * K + hi * 8;
  const unsigned short* wp2 = W + (size_t)(n0 + 32 + l16) * K + hi * 8;
  const unsigned short* wp3 = W + (size_t)(n0 + 48 + l16) * K + hi * 8;

  for (int k0 = 0; k0 < K; k0 += 32) {
    v16bf a0 = ldfrag(ap0 + k0);
    v16bf a1 = ldfrag(ap1 + k0);
    v16bf b0 = ldfrag(wp0 + k0);
    v16bf b1 = ldfrag(wp1 + k0);
    v16bf b2 = ldfrag(wp2 + k0);
    v16bf b3 = ldfrag(wp3 + k0);
    WMMA_BF16(acc[0][0], a0, b0);
    WMMA_BF16(acc[0][1], a0, b1);
    WMMA_BF16(acc[0][2], a0, b2);
    WMMA_BF16(acc[0][3], a0, b3);
    WMMA_BF16(acc[1][0], a1, b0);
    WMMA_BF16(acc[1][1], a1, b1);
    WMMA_BF16(acc[1][2], a1, b2);
    WMMA_BF16(acc[1][3], a1, b3);
  }

#pragma unroll
  for (int i = 0; i < 2; ++i) {
    const int row0 = m0 + i * 16 + hi * 8;
#pragma unroll
    for (int t = 0; t < 4; ++t) {
      const int col = n0 + t * 16 + l16;
      const float bb = bias ? bias[col] : 0.f;
#pragma unroll
      for (int r = 0; r < 8; ++r) {
        float v = acc[i][t][r] + bb;
        if (ACT == 1) v = fmaxf(v, 0.f);
        if (ACT == 2) v = gelu_exact(v);
        const size_t off = (size_t)(row0 + r) * N + col;
        if (OUTMODE == 1)
          ((unsigned short*)Cout)[off] = f2bf(v);
        else
          ((float*)Cout)[off] = v;
      }
    }
  }
}

// ------------- fused expert: comb += gate * (gelu(z@W1^T+b1) @ W2^T + b2) ----
// Block = 128 threads = 4 waves; block tile = 128 tokens; each wave owns 32
// token rows end-to-end. Hidden activations (128 x 512 bf16, padded stride
// 520) live entirely in LDS (133 KB of the WGP's 320 KB) -- no HBM round-trip.
__global__ __launch_bounds__(128) void moe_expert_fused_kernel(
    const unsigned short* __restrict__ Abf,                     // (NTOK,128) bf16
    const unsigned short* __restrict__ W1, const float* __restrict__ B1,
    const unsigned short* __restrict__ W2, const float* __restrict__ B2,
    const float* __restrict__ gatevec, int gstride, int gidx,
    float* __restrict__ comb) {                                 // (NTOK,128) +=
  extern __shared__ char smem[];
  unsigned short* ehs = (unsigned short*)smem;   // 128 x LDST_ bf16
  const int lane = threadIdx.x & 31;
  const int wave = threadIdx.x >> 5;             // 0..3
  const int l16  = lane & 15;
  const int hi   = lane >> 4;
  const int m0   = blockIdx.x * 128 + wave * 32; // global token row base
  const int lrow = wave * 32;                    // LDS row base

  const unsigned short* ap0 = Abf + (size_t)(m0 + l16) * D_ + hi * 8;
  const unsigned short* ap1 = Abf + (size_t)(m0 + 16 + l16) * D_ + hi * 8;

  // ---------- phase 1: eh = gelu(z @ W1^T + b1) -> LDS ----------
  for (int n0 = 0; n0 < HID_; n0 += 64) {
    v8f acc[2][4] = {};
    const unsigned short* wp0 = W1 + (size_t)(n0 +  0 + l16) * D_ + hi * 8;
    const unsigned short* wp1 = W1 + (size_t)(n0 + 16 + l16) * D_ + hi * 8;
    const unsigned short* wp2 = W1 + (size_t)(n0 + 32 + l16) * D_ + hi * 8;
    const unsigned short* wp3 = W1 + (size_t)(n0 + 48 + l16) * D_ + hi * 8;
#pragma unroll
    for (int k0 = 0; k0 < D_; k0 += 32) {
      v16bf a0 = ldfrag(ap0 + k0);
      v16bf a1 = ldfrag(ap1 + k0);
      v16bf b0 = ldfrag(wp0 + k0);
      v16bf b1 = ldfrag(wp1 + k0);
      v16bf b2 = ldfrag(wp2 + k0);
      v16bf b3 = ldfrag(wp3 + k0);
      WMMA_BF16(acc[0][0], a0, b0);
      WMMA_BF16(acc[0][1], a0, b1);
      WMMA_BF16(acc[0][2], a0, b2);
      WMMA_BF16(acc[0][3], a0, b3);
      WMMA_BF16(acc[1][0], a1, b0);
      WMMA_BF16(acc[1][1], a1, b1);
      WMMA_BF16(acc[1][2], a1, b2);
      WMMA_BF16(acc[1][3], a1, b3);
    }
#pragma unroll
    for (int i = 0; i < 2; ++i) {
      const int row = lrow + i * 16 + hi * 8;
#pragma unroll
      for (int t = 0; t < 4; ++t) {
        const int col = n0 + t * 16 + l16;
        const float bb = B1[col];
#pragma unroll
        for (int r = 0; r < 8; ++r) {
          float v = gelu_exact(acc[i][t][r] + bb);
          ehs[(size_t)(row + r) * LDST_ + col] = f2bf(v);
        }
      }
    }
  }
  __syncthreads();

  // ---------- phase 2: comb += gate * (eh @ W2^T + b2), eh from LDS ----------
  const unsigned short* lp0 = ehs + (size_t)(lrow + l16) * LDST_ + hi * 8;
  const unsigned short* lp1 = ehs + (size_t)(lrow + 16 + l16) * LDST_ + hi * 8;
  for (int nb = 0; nb < D_; nb += 64) {
    v8f acc[2][4] = {};
    const unsigned short* q0 = W2 + (size_t)(nb +  0 + l16) * HID_ + hi * 8;
    const unsigned short* q1 = W2 + (size_t)(nb + 16 + l16) * HID_ + hi * 8;
    const unsigned short* q2 = W2 + (size_t)(nb + 32 + l16) * HID_ + hi * 8;
    const unsigned short* q3 = W2 + (size_t)(nb + 48 + l16) * HID_ + hi * 8;
    for (int k0 = 0; k0 < HID_; k0 += 32) {
      v16bf a0 = ldfrag(lp0 + k0);
      v16bf a1 = ldfrag(lp1 + k0);
      v16bf b0 = ldfrag(q0 + k0);
      v16bf b1 = ldfrag(q1 + k0);
      v16bf b2 = ldfrag(q2 + k0);
      v16bf b3 = ldfrag(q3 + k0);
      WMMA_BF16(acc[0][0], a0, b0);
      WMMA_BF16(acc[0][1], a0, b1);
      WMMA_BF16(acc[0][2], a0, b2);
      WMMA_BF16(acc[0][3], a0, b3);
      WMMA_BF16(acc[1][0], a1, b0);
      WMMA_BF16(acc[1][1], a1, b1);
      WMMA_BF16(acc[1][2], a1, b2);
      WMMA_BF16(acc[1][3], a1, b3);
    }
#pragma unroll
    for (int i = 0; i < 2; ++i) {
      const int row0 = m0 + i * 16 + hi * 8;
#pragma unroll
      for (int t = 0; t < 4; ++t) {
        const int col = nb + t * 16 + l16;
        const float bb = B2[col];
#pragma unroll
        for (int r = 0; r < 8; ++r) {
          const int row = row0 + r;
          float v = acc[i][t][r] + bb;
          float g = gatevec[(size_t)row * gstride + gidx];
          comb[(size_t)row * D_ + col] += g * v;
        }
      }
    }
  }
}

// ---------------- attention (per batch,head), T=100, hd=32 ----------------
__global__ __launch_bounds__(128) void attn_kernel(const float* __restrict__ qkv,
                                                   float* __restrict__ out) {
  __shared__ float qs[T_][32];
  __shared__ float ks[T_][32];
  __shared__ float vs[T_][32];
  const int b = blockIdx.x, h = blockIdx.y;
  const int tid = threadIdx.x;
  for (int i = tid; i < T_ * 32; i += blockDim.x) {
    int t = i >> 5, d = i & 31;
    long base = (long)(b * T_ + t) * (3 * D_);
    qs[t][d] = qkv[base + h * 32 + d];
    ks[t][d] = qkv[base + D_ + h * 32 + d];
    vs[t][d] = qkv[base + 2 * D_ + h * 32 + d];
  }
  __syncthreads();
  if (tid >= T_) return;
  const float scale = 0.1767766952966369f;  // 1/sqrt(32)
  float sc[T_];
  float mx = -1e30f;
  for (int j = 0; j < T_; ++j) {
    float s = 0.f;
#pragma unroll
    for (int d = 0; d < 32; ++d) s += qs[tid][d] * ks[j][d];
    s *= scale;
    sc[j] = s;
    mx = fmaxf(mx, s);
  }
  float sum = 0.f;
  for (int j = 0; j < T_; ++j) {
    sc[j] = __expf(sc[j] - mx);
    sum += sc[j];
  }
  float inv = 1.f / sum;
#pragma unroll 4
  for (int d = 0; d < 32; ++d) {
    float acc = 0.f;
    for (int j = 0; j < T_; ++j) acc += sc[j] * vs[j][d];
    out[(long)(b * T_ + tid) * D_ + h * 32 + d] = acc * inv;
  }
}

// ---------------- z = LayerNorm(z + a) * s + b, row-wise over D=128 ----------
__global__ __launch_bounds__(128) void add_ln_kernel(float* __restrict__ z,
                                                     const float* __restrict__ a,
                                                     const float* __restrict__ s,
                                                     const float* __restrict__ b) {
  __shared__ float red[128];
  const int row = blockIdx.x;
  const int d = threadIdx.x;
  float v = z[(long)row * D_ + d] + a[(long)row * D_ + d];
  red[d] = v;
  __syncthreads();
  for (int off = 64; off > 0; off >>= 1) {
    if (d < off) red[d] += red[d + off];
    __syncthreads();
  }
  float mean = red[0] * (1.f / D_);
  __syncthreads();
  float c = v - mean;
  red[d] = c * c;
  __syncthreads();
  for (int off = 64; off > 0; off >>= 1) {
    if (d < off) red[d] += red[d + off];
    __syncthreads();
  }
  float var = red[0] * (1.f / D_);
  z[(long)row * D_ + d] = c * rsqrtf(var + 1e-5f) * s[d] + b[d];
}

// ---------------- specific router: softmax over 6, top-2 renormalized --------
__global__ void router_spec_kernel(const float* __restrict__ z,
                                   const float* __restrict__ wr,
                                   float* __restrict__ gates,
                                   float* __restrict__ probs_out) {
  int n = blockIdx.x * blockDim.x + threadIdx.x;
  if (n >= NTOK_) return;
  const float* x = z + (long)n * D_;
  float lg[ES_];
#pragma unroll
  for (int e = 0; e < ES_; ++e) {
    const float* w = wr + e * D_;
    float s = 0.f;
    for (int d = 0; d < D_; ++d) s += x[d] * w[d];
    lg[e] = s;
  }
  float mx = lg[0];
#pragma unroll
  for (int e = 1; e < ES_; ++e) mx = fmaxf(mx, lg[e]);
  float sum = 0.f;
#pragma unroll
  for (int e = 0; e < ES_; ++e) { lg[e] = __expf(lg[e] - mx); sum += lg[e]; }
  float inv = 1.f / sum;
  float p[ES_];
#pragma unroll
  for (int e = 0; e < ES_; ++e) {
    p[e] = lg[e] * inv;
    probs_out[(long)n * ES_ + e] = p[e];
  }
  int i1 = 0;
#pragma unroll
  for (int e = 1; e < ES_; ++e) if (p[e] > p[i1]) i1 = e;
  int i2 = (i1 == 0) ? 1 : 0;
#pragma unroll
  for (int e = 0; e < ES_; ++e) if (e != i1 && p[e] > p[i2]) i2 = e;
  float g = 1.f / (p[i1] + p[i2]);
#pragma unroll
  for (int e = 0; e < ES_; ++e) gates[(long)n * ES_ + e] = 0.f;
  gates[(long)n * ES_ + i1] = p[i1] * g;
  gates[(long)n * ES_ + i2] = p[i2] * g;
}

// ---------------- shared router: soft softmax over 2 -------------------------
__global__ void router_shr_kernel(const float* __restrict__ z,
                                  const float* __restrict__ wr,
                                  float* __restrict__ gatesS) {
  int n = blockIdx.x * blockDim.x + threadIdx.x;
  if (n >= NTOK_) return;
  const float* x = z + (long)n * D_;
  float lg[ESH_];
#pragma unroll
  for (int e = 0; e < ESH_; ++e) {
    const float* w = wr + e * D_;
    float s = 0.f;
    for (int d = 0; d < D_; ++d) s += x[d] * w[d];
    lg[e] = s;
  }
  float mx = fmaxf(lg[0], lg[1]);
  float e0 = __expf(lg[0] - mx), e1 = __expf(lg[1] - mx);
  float inv = 1.f / (e0 + e1);
  gatesS[(long)n * ESH_ + 0] = e0 * inv;
  gatesS[(long)n * ESH_ + 1] = e1 * inv;
}

// ---------------- z += pos_embed ---------------------------------------------
__global__ void add_pos_kernel(float* __restrict__ z, const float* __restrict__ pos) {
  long idx = (long)blockIdx.x * blockDim.x + threadIdx.x;
  if (idx >= (long)NTOK_ * D_) return;
  int d = (int)(idx % D_);
  long n = idx / D_;
  int t = (int)(n % T_);
  z[idx] += pos[t * D_ + d];
}

// ---------------- mean over T ------------------------------------------------
__global__ void mean_kernel(const float* __restrict__ z, float* __restrict__ meanb) {
  int idx = blockIdx.x * blockDim.x + threadIdx.x;
  if (idx >= B_ * D_) return;
  int b = idx / D_, d = idx % D_;
  float s = 0.f;
  for (int t = 0; t < T_; ++t) s += z[(long)(b * T_ + t) * D_ + d];
  meanb[idx] = s * (1.f / T_);
}

// ---------------- classifier -------------------------------------------------
__global__ void cls_kernel(const float* __restrict__ meanb,
                           const float* __restrict__ w,
                           const float* __restrict__ bias,
                           float* __restrict__ out) {
  int idx = blockIdx.x * blockDim.x + threadIdx.x;
  if (idx >= B_ * NC_) return;
  int b = idx / NC_, c = idx % NC_;
  float s = bias[c];
  for (int d = 0; d < D_; ++d) s += meanb[b * D_ + d] * w[c * D_ + d];
  out[idx] = s;
}

// =============================================================================
extern "C" void kernel_launch(void* const* d_in, const int* in_sizes, int n_in,
                              void* d_out, int out_size, void* d_ws, size_t ws_size,
                              hipStream_t stream) {
  (void)in_sizes; (void)n_in; (void)out_size; (void)ws_size;
  const float* x          = (const float*)d_in[0];
  const float* sf_w1      = (const float*)d_in[1];
  const float* sf_b1      = (const float*)d_in[2];
  const float* sf_w2      = (const float*)d_in[3];
  const float* sf_b2      = (const float*)d_in[4];
  const float* pos        = (const float*)d_in[5];
  const float* attn_wqkv  = (const float*)d_in[6];
  const float* attn_bqkv  = (const float*)d_in[7];
  const float* attn_wo    = (const float*)d_in[8];
  const float* attn_bo    = (const float*)d_in[9];
  const float* ln1_s      = (const float*)d_in[10];
  const float* ln1_b      = (const float*)d_in[11];
  const float* spec_router= (const float*)d_in[12];
  const float* spec_w1    = (const float*)d_in[13];
  const float* spec_b1    = (const float*)d_in[14];
  const float* spec_w2    = (const float*)d_in[15];
  const float* spec_b2    = (const float*)d_in[16];
  const float* shr_router = (const float*)d_in[17];
  const float* shr_w1     = (const float*)d_in[18];
  const float* shr_b1     = (const float*)d_in[19];
  const float* shr_w2     = (const float*)d_in[20];
  const float* shr_b2     = (const float*)d_in[21];
  const float* ln2_s      = (const float*)d_in[22];
  const float* ln2_b      = (const float*)d_in[23];
  const float* cls_w      = (const float*)d_in[24];
  const float* cls_b      = (const float*)d_in[25];

  float* logits_out = (float*)d_out;
  float* probs_out  = (float*)d_out + (size_t)B_ * NC_;

  // ---- workspace bump allocator ----
  char* wp = (char*)d_ws;
  auto alloc = [&](size_t bytes) -> void* {
    void* r = (void*)wp;
    wp += (bytes + 255) & ~(size_t)255;
    return r;
  };
  unsigned short* xbf    = (unsigned short*)alloc((size_t)NTOK_ * FLATP_ * 2);
  unsigned short* h1bf   = (unsigned short*)alloc((size_t)NTOK_ * ENCH_ * 2);
  float*          z      = (float*)alloc((size_t)NTOK_ * D_ * 4);
  unsigned short* zbf    = (unsigned short*)alloc((size_t)NTOK_ * D_ * 2);
  float*          qkv    = (float*)alloc((size_t)NTOK_ * 3 * D_ * 4);
  float*          attno  = (float*)alloc((size_t)NTOK_ * D_ * 4);
  unsigned short* attnobf= (unsigned short*)alloc((size_t)NTOK_ * D_ * 2);
  float*          proj   = (float*)alloc((size_t)NTOK_ * D_ * 4);
  float*          comb   = (float*)alloc((size_t)NTOK_ * D_ * 4);
  float*          gates  = (float*)alloc((size_t)NTOK_ * ES_ * 4);
  float*          gatesS = (float*)alloc((size_t)NTOK_ * ESH_ * 4);
  float*          meanb  = (float*)alloc((size_t)B_ * D_ * 4);
  unsigned short* wbuf   = (unsigned short*)alloc((size_t)HID_ * D_ * 2);
  unsigned short* wbuf1  = (unsigned short*)alloc((size_t)HID_ * D_ * 2);
  unsigned short* wbuf2  = (unsigned short*)alloc((size_t)HID_ * D_ * 2);

  auto cvt = [&](const float* src, unsigned short* dst, int M, int K, int Kp) {
    long tot = (long)M * Kp;
    int blocks = (int)((tot + 255) / 256);
    cvt_pad_bf16_kernel<<<blocks, 256, 0, stream>>>(src, dst, M, K, Kp);
  };
  const int ND_BLOCKS = (int)(((long)NTOK_ * D_ + 255) / 256);
  const int TOK_BLOCKS = (NTOK_ + 255) / 256;
  const int GY = NTOK_ / 256;           // standalone GEMM grid rows
  const int FGRID = NTOK_ / 128;        // fused expert grid (200 blocks)
  const size_t FUSED_LDS = (size_t)128 * LDST_ * 2;  // 133,120 B of 320 KB/WGP

  // fused-expert launcher
  auto expert = [&](const float* w1, const float* b1, const float* w2,
                    const float* b2, size_t idx, const float* gv, int gstride,
                    int gidx) {
    cvt(w1 + idx * HID_ * D_, wbuf1, HID_, D_, D_);
    cvt(w2 + idx * D_ * HID_, wbuf2, D_, HID_, HID_);
    moe_expert_fused_kernel<<<FGRID, 128, FUSED_LDS, stream>>>(
        zbf, wbuf1, b1 + idx * HID_, wbuf2, b2 + idx * D_, gv, gstride, gidx,
        comb);
  };

  // ================= stem =================
  cvt(x, xbf, NTOK_, FLAT_, FLATP_);
  cvt(sf_w1, wbuf, ENCH_, FLAT_, FLATP_);
  gemm_bf16_wmma<1, 1><<<dim3(ENCH_ / 64, GY), 256, 0, stream>>>(
      xbf, wbuf, sf_b1, h1bf, NTOK_, ENCH_, FLATP_);
  cvt(sf_w2, wbuf, D_, ENCH_, ENCH_);
  gemm_bf16_wmma<1, 0><<<dim3(D_ / 64, GY), 256, 0, stream>>>(
      h1bf, wbuf, sf_b2, z, NTOK_, D_, ENCH_);
  add_pos_kernel<<<ND_BLOCKS, 256, 0, stream>>>(z, pos);

  // ================= transformer layers =================
  for (int l = 0; l < L_; ++l) {
    // --- attention ---
    cvt(z, zbf, NTOK_, D_, D_);
    cvt(attn_wqkv + (size_t)l * 3 * D_ * D_, wbuf, 3 * D_, D_, D_);
    gemm_bf16_wmma<0, 0><<<dim3(3 * D_ / 64, GY), 256, 0, stream>>>(
        zbf, wbuf, attn_bqkv + (size_t)l * 3 * D_, qkv, NTOK_, 3 * D_, D_);
    attn_kernel<<<dim3(B_, NH_), 128, 0, stream>>>(qkv, attno);
    cvt(attno, attnobf, NTOK_, D_, D_);
    cvt(attn_wo + (size_t)l * D_ * D_, wbuf, D_, D_, D_);
    gemm_bf16_wmma<0, 0><<<dim3(D_ / 64, GY), 256, 0, stream>>>(
        attnobf, wbuf, attn_bo + (size_t)l * D_, proj, NTOK_, D_, D_);
    add_ln_kernel<<<NTOK_, 128, 0, stream>>>(z, proj, ln1_s + (size_t)l * D_,
                                             ln1_b + (size_t)l * D_);
    // --- routing ---
    cvt(z, zbf, NTOK_, D_, D_);
    router_spec_kernel<<<TOK_BLOCKS, 256, 0, stream>>>(
        z, spec_router + (size_t)l * ES_ * D_, gates,
        probs_out + (size_t)l * NTOK_ * ES_);
    router_shr_kernel<<<TOK_BLOCKS, 256, 0, stream>>>(
        z, shr_router + (size_t)l * ESH_ * D_, gatesS);
    hipMemsetAsync(comb, 0, (size_t)NTOK_ * D_ * 4, stream);
    // --- specific experts (fused GEMM1+GELU+GEMM2, eh resident in LDS) ---
    for (int e = 0; e < ES_; ++e)
      expert(spec_w1, spec_b1, spec_w2, spec_b2, (size_t)l * ES_ + e, gates,
             ES_, e);
    // --- shared experts ---
    for (int e = 0; e < ESH_; ++e)
      expert(shr_w1, shr_b1, shr_w2, shr_b2, (size_t)l * ESH_ + e, gatesS,
             ESH_, e);
    add_ln_kernel<<<NTOK_, 128, 0, stream>>>(z, comb, ln2_s + (size_t)l * D_,
                                             ln2_b + (size_t)l * D_);
  }

  // ================= head =================
  mean_kernel<<<(B_ * D_ + 255) / 256, 256, 0, stream>>>(z, meanb);
  cls_kernel<<<(B_ * NC_ + 63) / 64, 64, 0, stream>>>(meanb, cls_w, cls_b, logits_out);
}
